// BertBiLSTMCRF_52493090291984
// MI455X (gfx1250) — matrix operand order repaired and checked by reading, not compiled, not run on hardware
//
#include <hip/hip_runtime.h>
#include <hip/hip_bf16.h>
#include <math.h>

// ---------------- problem constants (from reference) ----------------
static constexpr int kB  = 4;
static constexpr int kS  = 512;
static constexpr int kH  = 768;
static constexpr int kNL = 12;
static constexpr int kNH = 12;
static constexpr int kFF = 3072;
static constexpr int kNT = 25;
static constexpr int kLH = 256;
static constexpr int kHD = 64;       // H / NH
static constexpr int kG4 = 4 * kLH;  // 1024
static constexpr int kD2 = 2 * kLH;  // 512
static constexpr int kRows = kB * kS; // 2048

// ---------------- WMMA types ----------------
typedef __bf16 bf16x16 __attribute__((ext_vector_type(16)));
typedef float  floatx8 __attribute__((ext_vector_type(8)));

union WFrag {
  unsigned int u[8];
  bf16x16 v;
};

__device__ __forceinline__ unsigned short f32_to_bf16_bits(float f) {
  union { float f; unsigned int u; } a;
  a.f = f;
  unsigned int u = a.u;
  unsigned int r = u + 0x7FFFu + ((u >> 16) & 1u); // round to nearest even
  return (unsigned short)(r >> 16);
}

__device__ __forceinline__ unsigned int pack_bf16x2(float lo, float hi) {
  return (unsigned int)f32_to_bf16_bits(lo) |
         ((unsigned int)f32_to_bf16_bits(hi) << 16);
}

enum { EPI_NONE = 0, EPI_BIAS = 1, EPI_BIAS_GELU = 2 };

// LDS tile row stride (bf16 elements): 40 -> 80B rows (8B aligned),
// 20-bank stride keeps 16-row fragment reads conflict-free.
static constexpr int kTS = 40;

// ---------------- bf16 WMMA GEMM: C = A(MxK) * B(KxN) [+bias][gelu] ----------
// A row-major f32 (lda).  B row-major [K,N] (ldb=N) or, if TRANSB, [N,K]
// (ldb=K).  C row-major f32 (ldc).  FULL => M%128==0, N%128==0, K%32==0 and
// 16B-alignable loads (no bounds checks, vectorized staging).
template <bool TRANSB, bool FULL>
__global__ __launch_bounds__(256)
void gemm_wmma_bf16(const float* __restrict__ A, const float* __restrict__ Bm,
                    const float* __restrict__ bias, float* __restrict__ C,
                    int M, int N, int K, int lda, int ldb, int ldc, int epi)
{
  __shared__ __align__(16) unsigned short As[128][kTS]; // [m][k]
  __shared__ __align__(16) unsigned short Bs[128][kTS]; // [n][k]

  const int tid  = threadIdx.x;
  const int lane = tid & 31;
  const int wid  = tid >> 5;
  const int wm   = wid >> 1;   // 0..3 -> 32 rows each
  const int wn   = wid & 1;    // 0..1 -> 64 cols each
  const int m0   = blockIdx.x * 128;
  const int n0   = blockIdx.y * 128;

  floatx8 acc[2][4];
#pragma unroll
  for (int i = 0; i < 2; ++i)
#pragma unroll
    for (int j = 0; j < 4; ++j)
#pragma unroll
      for (int r = 0; r < 8; ++r) acc[i][j][r] = 0.0f;

  for (int k0 = 0; k0 < K; k0 += 32) {
    if (FULL) {
      // ---- issue ALL tile loads first (keep VMEM latency overlapped) ----
      float4 av[4];
#pragma unroll
      for (int i = 0; i < 4; ++i) {
        int idx = tid + i * 256;
        int r  = idx >> 3;
        int kq = (idx & 7) * 4;
        av[i] = *(const float4*)&A[(long)(m0 + r) * lda + k0 + kq];
      }
      if (TRANSB) {
        float4 bv[4];
#pragma unroll
        for (int i = 0; i < 4; ++i) {
          int idx = tid + i * 256;
          int n  = idx >> 3;
          int kq = (idx & 7) * 4;
          bv[i] = *(const float4*)&Bm[(long)(n0 + n) * ldb + k0 + kq];
        }
        // prefetch next K tile while packing
        if (k0 + 32 < K) {
          __builtin_prefetch(&A[(long)(m0 + (tid >> 1)) * lda + k0 + 32 + (tid & 1) * 16], 0, 1);
          __builtin_prefetch(&Bm[(long)(n0 + (tid >> 1)) * ldb + k0 + 32 + (tid & 1) * 16], 0, 1);
        }
        // ---- pack + store ----
#pragma unroll
        for (int i = 0; i < 4; ++i) {
          int idx = tid + i * 256;
          int r  = idx >> 3;
          int kq = (idx & 7) * 4;
          uint2 p;
          p.x = pack_bf16x2(av[i].x, av[i].y);
          p.y = pack_bf16x2(av[i].z, av[i].w);
          *(uint2*)&As[r][kq] = p;
        }
#pragma unroll
        for (int i = 0; i < 4; ++i) {
          int idx = tid + i * 256;
          int n  = idx >> 3;
          int kq = (idx & 7) * 4;
          uint2 p;
          p.x = pack_bf16x2(bv[i].x, bv[i].y);
          p.y = pack_bf16x2(bv[i].z, bv[i].w);
          *(uint2*)&Bs[n][kq] = p;
        }
      } else {
        // B is [K,N]: float4 along n from two adjacent k rows, pack (k,k+1)
        float4 b0[2], b1[2];
#pragma unroll
        for (int i = 0; i < 2; ++i) {
          int idx = tid + i * 256;          // 0..511
          int n4 = (idx & 31) * 4;
          int kk = (idx >> 5) * 2;          // even
          b0[i] = *(const float4*)&Bm[(long)(k0 + kk) * ldb + n0 + n4];
          b1[i] = *(const float4*)&Bm[(long)(k0 + kk + 1) * ldb + n0 + n4];
        }
        if (k0 + 32 < K) {
          __builtin_prefetch(&A[(long)(m0 + (tid >> 1)) * lda + k0 + 32 + (tid & 1) * 16], 0, 1);
          __builtin_prefetch(&Bm[(long)(k0 + 32 + (tid >> 3)) * ldb + n0 + (tid & 7) * 16], 0, 1);
        }
#pragma unroll
        for (int i = 0; i < 4; ++i) {
          int idx = tid + i * 256;
          int r  = idx >> 3;
          int kq = (idx & 7) * 4;
          uint2 p;
          p.x = pack_bf16x2(av[i].x, av[i].y);
          p.y = pack_bf16x2(av[i].z, av[i].w);
          *(uint2*)&As[r][kq] = p;
        }
#pragma unroll
        for (int i = 0; i < 2; ++i) {
          int idx = tid + i * 256;
          int n4 = (idx & 31) * 4;
          int kk = (idx >> 5) * 2;
          *(unsigned int*)&Bs[n4 + 0][kk] = pack_bf16x2(b0[i].x, b1[i].x);
          *(unsigned int*)&Bs[n4 + 1][kk] = pack_bf16x2(b0[i].y, b1[i].y);
          *(unsigned int*)&Bs[n4 + 2][kk] = pack_bf16x2(b0[i].z, b1[i].z);
          *(unsigned int*)&Bs[n4 + 3][kk] = pack_bf16x2(b0[i].w, b1[i].w);
        }
      }
    } else {
      // ---- guarded scalar staging (only the tiny N=25 head GEMM) ----
#pragma unroll
      for (int i = 0; i < 16; ++i) {
        int idx = tid + i * 256;
        int r = idx >> 5, kk = idx & 31;
        int gm = m0 + r, gk = k0 + kk;
        float v = (gm < M && gk < K) ? A[(long)gm * lda + gk] : 0.0f;
        As[r][kk] = f32_to_bf16_bits(v);
      }
#pragma unroll
      for (int i = 0; i < 16; ++i) {
        int idx = tid + i * 256;
        int n, kk;
        if (TRANSB) { n = idx >> 5; kk = idx & 31; }
        else        { n = idx & 127; kk = idx >> 7; }
        int gn = n0 + n, gk = k0 + kk;
        float v;
        if (TRANSB) v = (gn < N && gk < K) ? Bm[(long)gn * ldb + gk] : 0.0f;
        else        v = (gn < N && gk < K) ? Bm[(long)gk * ldb + gn] : 0.0f;
        Bs[n][kk] = f32_to_bf16_bits(v);
      }
    }
    __syncthreads();

    // ---- build fragments per ISA 16-bit layouts ----
    // A 16x32: lanes 0-15 row=lane, K pairs {0..7,16..23}; lanes 16-31 +8
    WFrag a[2], b[4];
    const int arow  = lane & 15;
    const int akoff = (lane >> 4) * 8;
#pragma unroll
    for (int mi = 0; mi < 2; ++mi) {
      int r = wm * 32 + mi * 16 + arow;
#pragma unroll
      for (int vv = 0; vv < 8; ++vv) {
        int kk = ((vv & 3) * 2) + ((vv >> 2) * 16) + akoff;
        a[mi].u[vv] = *(const unsigned int*)&As[r][kk];
      }
    }
    // B 32x16: lanes 0-15 N=lane, K pairs {0..7,16..23}; lanes 16-31 +8
    const int bnl   = lane & 15;
    const int bkoff = (lane >> 4) * 8;
#pragma unroll
    for (int ni = 0; ni < 4; ++ni) {
      int n = wn * 64 + ni * 16 + bnl;
#pragma unroll
      for (int vv = 0; vv < 8; ++vv) {
        b[ni].u[vv] = *(const unsigned int*)&Bs[n][((vv & 3) * 2) + ((vv >> 2) * 16) + bkoff];
      }
    }
#pragma unroll
    for (int mi = 0; mi < 2; ++mi)
#pragma unroll
      for (int ni = 0; ni < 4; ++ni)
        acc[mi][ni] = __builtin_amdgcn_wmma_f32_16x16x32_bf16(
            false, a[mi].v, false, b[ni].v, (short)0, acc[mi][ni], false, false);
    __syncthreads();
  }

  // ---- epilogue: C/D layout: lane0-15 n=lane, vgpr r -> m=r (+8 hi-half) ----
  const int mlane = (lane >> 4) * 8;
  const int nlane = lane & 15;
#pragma unroll
  for (int mi = 0; mi < 2; ++mi) {
#pragma unroll
    for (int ni = 0; ni < 4; ++ni) {
#pragma unroll
      for (int r = 0; r < 8; ++r) {
        int m = m0 + wm * 32 + mi * 16 + mlane + r;
        int n = n0 + wn * 64 + ni * 16 + nlane;
        if (FULL || (m < M && n < N)) {
          float v = acc[mi][ni][r];
          if (epi != EPI_NONE) v += bias[n];
          if (epi == EPI_BIAS_GELU)
            v = 0.5f * v * (1.0f + erff(v * 0.70710678118654752f));
          C[(long)m * ldc + n] = v;
        }
      }
    }
  }
}

// ---------------- block reduction (256 threads) ----------------
__device__ __forceinline__ float blockReduceSum256(float v) {
  __shared__ float red[8];
  __shared__ float bc;
  for (int off = 16; off > 0; off >>= 1) v += __shfl_down(v, off);
  int lane = threadIdx.x & 31, wid = threadIdx.x >> 5;
  if (lane == 0) red[wid] = v;
  __syncthreads();
  v = (threadIdx.x < 8) ? red[threadIdx.x] : 0.0f;
  if (wid == 0)
    for (int off = 4; off > 0; off >>= 1) v += __shfl_down(v, off);
  if (threadIdx.x == 0) bc = v;
  __syncthreads();
  return bc;
}

// ---------------- embedding + LayerNorm ----------------
__global__ __launch_bounds__(256)
void embed_ln_kernel(const int* __restrict__ ids, const float* __restrict__ wemb,
                     const float* __restrict__ pemb, const float* __restrict__ temb,
                     const float* __restrict__ g, const float* __restrict__ be,
                     float* __restrict__ x)
{
  long row = blockIdx.x;                  // b*S + s
  int s = (int)(row % kS);
  int id = ids[row];
  float loc[3]; float lsum = 0.0f;
#pragma unroll
  for (int i = 0; i < 3; ++i) {
    int h = threadIdx.x + i * 256;
    float v = wemb[(long)id * kH + h] + pemb[(long)s * kH + h] + temb[h];
    loc[i] = v; lsum += v;
  }
  float mean = blockReduceSum256(lsum) * (1.0f / kH);
  float lvar = 0.0f;
#pragma unroll
  for (int i = 0; i < 3; ++i) { float d = loc[i] - mean; lvar += d * d; }
  float var = blockReduceSum256(lvar) * (1.0f / kH);
  float rstd = rsqrtf(var + 1e-12f);
#pragma unroll
  for (int i = 0; i < 3; ++i) {
    int h = threadIdx.x + i * 256;
    x[row * kH + h] = (loc[i] - mean) * rstd * g[h] + be[h];
  }
}

// ---------------- residual + LayerNorm: x = LN(x + t) ----------------
__global__ __launch_bounds__(256)
void ln_residual_kernel(float* __restrict__ x, const float* __restrict__ t,
                        const float* __restrict__ g, const float* __restrict__ be)
{
  long row = blockIdx.x;
  float loc[3]; float lsum = 0.0f;
#pragma unroll
  for (int i = 0; i < 3; ++i) {
    int h = threadIdx.x + i * 256;
    float v = x[row * kH + h] + t[row * kH + h];
    loc[i] = v; lsum += v;
  }
  float mean = blockReduceSum256(lsum) * (1.0f / kH);
  float lvar = 0.0f;
#pragma unroll
  for (int i = 0; i < 3; ++i) { float d = loc[i] - mean; lvar += d * d; }
  float var = blockReduceSum256(lvar) * (1.0f / kH);
  float rstd = rsqrtf(var + 1e-12f);
#pragma unroll
  for (int i = 0; i < 3; ++i) {
    int h = threadIdx.x + i * 256;
    x[row * kH + h] = (loc[i] - mean) * rstd * g[h] + be[h];
  }
}

// ---------------- attention: wave per query row ----------------
__global__ __launch_bounds__(128)
void attention_kernel(const float* __restrict__ q, const float* __restrict__ kM,
                      const float* __restrict__ vM, const int* __restrict__ mask,
                      float* __restrict__ ctx)
{
  __shared__ float sc[4][kS];
  __shared__ float qs[4][kHD];
  const int w = threadIdx.x >> 5, lane = threadIdx.x & 31;
  const int gid = blockIdx.x * 4 + w;
  const int b  = gid / (kNH * kS);
  const int rem = gid % (kNH * kS);
  const int h  = rem / kS;
  const int qi = rem % kS;

  const float* qrow = q + ((long)(b * kS + qi) * kH) + h * kHD;
  float2 qv = ((const float2*)qrow)[lane];
  qs[w][2 * lane] = qv.x; qs[w][2 * lane + 1] = qv.y;

  float mmax = -1e30f;
  for (int kk = lane; kk < kS; kk += 32) {
    const float* krow = kM + ((long)(b * kS + kk) * kH) + h * kHD;
    float dot = 0.0f;
#pragma unroll 8
    for (int d = 0; d < kHD; d += 2) {
      float2 kv = *(const float2*)&krow[d];
      dot += qs[w][d] * kv.x + qs[w][d + 1] * kv.y;
    }
    float sbias = (1.0f - (float)mask[b * kS + kk]) * -1e9f;
    float s = dot * 0.125f + sbias;   // 1/sqrt(64)
    sc[w][kk] = s;
    mmax = fmaxf(mmax, s);
  }
  for (int off = 16; off > 0; off >>= 1) mmax = fmaxf(mmax, __shfl_xor(mmax, off));
  float ssum = 0.0f;
  for (int kk = lane; kk < kS; kk += 32) {
    float p = __expf(sc[w][kk] - mmax);
    sc[w][kk] = p; ssum += p;
  }
  for (int off = 16; off > 0; off >>= 1) ssum += __shfl_xor(ssum, off);
  float inv = 1.0f / ssum;
  float a0 = 0.0f, a1 = 0.0f;
  for (int kk = 0; kk < kS; ++kk) {
    float p = sc[w][kk];
    const float* vrow = vM + ((long)(b * kS + kk) * kH) + h * kHD;
    float2 vv = ((const float2*)vrow)[lane];
    a0 += p * vv.x; a1 += p * vv.y;
  }
  float* crow = ctx + ((long)(b * kS + qi) * kH) + h * kHD;
  ((float2*)crow)[lane] = make_float2(a0 * inv, a1 * inv);
}

// ---------------- generic [D0,D1,W] -> [D1,D0,W] transpose ----------------
__global__ void transpose01_kernel(const float* __restrict__ in, float* __restrict__ out,
                                   int D0, int D1, int W)
{
  long total = (long)D0 * D1 * W;
  for (long i = (long)blockIdx.x * blockDim.x + threadIdx.x; i < total;
       i += (long)gridDim.x * blockDim.x) {
    int w = (int)(i % W);
    long r = i / W;
    int d1 = (int)(r % D1);
    int d0 = (int)(r / D1);
    out[((long)d1 * D0 + d0) * W + w] = in[i];
  }
}

// ---------------- LSTM recurrence: one block per direction ----------------
// Gx*: [S,B,4*LH] = x@Wih^T + b (precomputed).  out: [S,B,2*LH].
__global__ __launch_bounds__(1024)
void lstm_rec_kernel(const float* __restrict__ GxF, const float* __restrict__ GxB,
                     const float* __restrict__ whhF, const float* __restrict__ whhB,
                     float* __restrict__ out)
{
  const int dir = blockIdx.x;            // 0 = forward, 1 = backward
  const float* Gx  = dir ? GxB  : GxF;
  const float* whh = dir ? whhB : whhF;
  const int coff = dir ? kLH : 0;

  __shared__ float hS[kB * kLH];
  __shared__ float cS[kB * kLH];
  __shared__ float gS[kB * kG4];

  const int tid = threadIdx.x;           // 0..1023
  hS[tid] = 0.0f; cS[tid] = 0.0f;
  __syncthreads();

  for (int step = 0; step < kS; ++step) {
    int t = dir ? (kS - 1 - step) : step;
    // phase A: gates[bi][j] = Gx + h @ Whh^T   (j = tid)
    const float* wr = whh + (long)tid * kLH;
    for (int bi = 0; bi < kB; ++bi) {
      float s = Gx[((long)t * kB + bi) * kG4 + tid];
      const float* hb = hS + bi * kLH;
      for (int kk = 0; kk < kLH; ++kk) s += hb[kk] * wr[kk];
      gS[bi * kG4 + tid] = s;
    }
    __syncthreads();
    // phase B: cell update (tid -> (bi,u))
    int bi = tid >> 8, u = tid & 255;
    float gi = gS[bi * kG4 + u];
    float gf = gS[bi * kG4 + kLH + u];
    float gg = gS[bi * kG4 + 2 * kLH + u];
    float go = gS[bi * kG4 + 3 * kLH + u];
    float si = 1.0f / (1.0f + __expf(-gi));
    float sf = 1.0f / (1.0f + __expf(-gf));
    float so = 1.0f / (1.0f + __expf(-go));
    float c  = sf * cS[bi * kLH + u] + si * tanhf(gg);
    float hh = so * tanhf(c);
    cS[bi * kLH + u] = c;
    hS[bi * kLH + u] = hh;
    out[((long)t * kB + bi) * kD2 + coff + u] = hh;
    __syncthreads();
  }
}

// ---------------- CRF negative log-likelihood ----------------
__global__ __launch_bounds__(128)
void crf_nll_kernel(const float* __restrict__ e /*[S,B,NT]*/,
                    const int* __restrict__ labels /*[B,S]*/,
                    const int* __restrict__ mask /*[B,S]*/,
                    const float* __restrict__ tr, const float* __restrict__ sv,
                    const float* __restrict__ ev, float* __restrict__ loss_out)
{
  __shared__ float alpha[2][kB][kNT];
  __shared__ float scores[kB];
  const int tid = threadIdx.x;
  const int b = tid / kNT, j = tid % kNT;

  if (tid < kB) {
    int bb = tid;
    int t0 = labels[bb * kS]; if (t0 == -100) t0 = 0;
    float sc = sv[t0] + e[(long)bb * kNT + t0];
    int prev = t0;
    for (int t = 1; t < kS; ++t) {
      int tt = labels[bb * kS + t]; if (tt == -100) tt = 0;
      float mt = (float)mask[bb * kS + t];
      sc += (tr[prev * kNT + tt] + e[((long)t * kB + bb) * kNT + tt]) * mt;
      if (mt > 0.0f) prev = tt;
    }
    sc += ev[prev];
    scores[bb] = sc;
  }
  if (tid < kB * kNT) alpha[0][b][j] = sv[j] + e[(long)b * kNT + j];
  __syncthreads();

  int cur = 0;
  for (int t = 1; t < kS; ++t) {
    if (tid < kB * kNT) {
      float mt = (float)mask[b * kS + t];
      float m = -1e30f;
      for (int i = 0; i < kNT; ++i)
        m = fmaxf(m, alpha[cur][b][i] + tr[i * kNT + j]);
      float ssum = 0.0f;
      for (int i = 0; i < kNT; ++i)
        ssum += __expf(alpha[cur][b][i] + tr[i * kNT + j] - m);
      float nw = m + logf(ssum) + e[((long)t * kB + b) * kNT + j];
      alpha[1 - cur][b][j] = (mt > 0.0f) ? nw : alpha[cur][b][j];
    }
    __syncthreads();
    cur = 1 - cur;
  }
  if (tid == 0) {
    float total = 0.0f;
    for (int bb = 0; bb < kB; ++bb) {
      float m = -1e30f;
      for (int jj = 0; jj < kNT; ++jj) m = fmaxf(m, alpha[cur][bb][jj] + ev[jj]);
      float ssum = 0.0f;
      for (int jj = 0; jj < kNT; ++jj) ssum += __expf(alpha[cur][bb][jj] + ev[jj] - m);
      float logZ = m + logf(ssum);
      total += -(scores[bb] - logZ);
    }
    loss_out[0] = total * (1.0f / kB);
  }
}

// ---------------- Viterbi decode ----------------
__global__ __launch_bounds__(128)
void viterbi_kernel(const float* __restrict__ e /*[S,B,NT]*/,
                    const float* __restrict__ tr, const float* __restrict__ sv,
                    const float* __restrict__ ev, int* __restrict__ bp,
                    float* __restrict__ preds /*[B,S] as floats*/)
{
  __shared__ float alpha[2][kB][kNT];
  const int tid = threadIdx.x;
  const int b = tid / kNT, j = tid % kNT;
  if (tid < kB * kNT) alpha[0][b][j] = sv[j] + e[(long)b * kNT + j];
  __syncthreads();
  int cur = 0;
  for (int t = 1; t < kS; ++t) {
    if (tid < kB * kNT) {
      float best = -1e30f; int bi = 0;
      for (int i = 0; i < kNT; ++i) {
        float v = alpha[cur][b][i] + tr[i * kNT + j];
        if (v > best) { best = v; bi = i; }
      }
      alpha[1 - cur][b][j] = best + e[((long)t * kB + b) * kNT + j];
      bp[((long)(t - 1) * kB + b) * kNT + j] = bi;
    }
    __syncthreads();
    cur = 1 - cur;
  }
  if (tid < kB) {
    int bb = tid;
    float best = -1e30f; int tag = 0;
    for (int jj = 0; jj < kNT; ++jj) {
      float v = alpha[cur][bb][jj] + ev[jj];
      if (v > best) { best = v; tag = jj; }
    }
    for (int t = kS - 1; t >= 1; --t) {
      preds[bb * kS + t] = (float)tag;
      tag = bp[((long)(t - 1) * kB + bb) * kNT + tag];
    }
    preds[bb * kS + 0] = (float)tag;
  }
}

// ---------------- host-side orchestration ----------------
static inline void launch_gemm(const float* A, const float* Bm, const float* bias,
                               float* C, int M, int N, int K, int lda, int ldb,
                               int ldc, int transB, int epi, hipStream_t stream)
{
  dim3 grid((M + 127) / 128, (N + 127) / 128);
  const bool full = (M % 128 == 0) && (N % 128 == 0) && (K % 32 == 0) &&
                    (lda % 4 == 0) && (ldb % 4 == 0);
  if (full) {
    if (transB)
      gemm_wmma_bf16<true, true><<<grid, 256, 0, stream>>>(A, Bm, bias, C, M, N, K,
                                                           lda, ldb, ldc, epi);
    else
      gemm_wmma_bf16<false, true><<<grid, 256, 0, stream>>>(A, Bm, bias, C, M, N, K,
                                                            lda, ldb, ldc, epi);
  } else {
    if (transB)
      gemm_wmma_bf16<true, false><<<grid, 256, 0, stream>>>(A, Bm, bias, C, M, N, K,
                                                            lda, ldb, ldc, epi);
    else
      gemm_wmma_bf16<false, false><<<grid, 256, 0, stream>>>(A, Bm, bias, C, M, N, K,
                                                             lda, ldb, ldc, epi);
  }
}

extern "C" void kernel_launch(void* const* d_in, const int* in_sizes, int n_in,
                              void* d_out, int out_size, void* d_ws, size_t ws_size,
                              hipStream_t stream)
{
  (void)in_sizes; (void)n_in; (void)out_size; (void)ws_size;

  const int* ids     = (const int*)d_in[0];
  const int* mask    = (const int*)d_in[1];
  const int* labels  = (const int*)d_in[2];
  const float* word_emb = (const float*)d_in[3];
  const float* pos_emb  = (const float*)d_in[4];
  const float* type_emb = (const float*)d_in[5];
  const float* emb_g    = (const float*)d_in[6];
  const float* emb_b    = (const float*)d_in[7];
  const float* wq = (const float*)d_in[8];   const float* bq = (const float*)d_in[9];
  const float* wk = (const float*)d_in[10];  const float* bk = (const float*)d_in[11];
  const float* wv = (const float*)d_in[12];  const float* bv = (const float*)d_in[13];
  const float* wo = (const float*)d_in[14];  const float* bo = (const float*)d_in[15];
  const float* ln1g = (const float*)d_in[16]; const float* ln1b = (const float*)d_in[17];
  const float* wi = (const float*)d_in[18];  const float* bi = (const float*)d_in[19];
  const float* wo2 = (const float*)d_in[20]; const float* bo2 = (const float*)d_in[21];
  const float* ln2g = (const float*)d_in[22]; const float* ln2b = (const float*)d_in[23];
  const float* head_w = (const float*)d_in[24];
  const float* head_b = (const float*)d_in[25];
  const float* trans  = (const float*)d_in[26];
  const float* startv = (const float*)d_in[27];
  const float* endv   = (const float*)d_in[28];
  // order: l0f, l0b, l1f, l1b
  const float* wih[4]  = {(const float*)d_in[29], (const float*)d_in[32],
                          (const float*)d_in[35], (const float*)d_in[38]};
  const float* whh[4]  = {(const float*)d_in[30], (const float*)d_in[33],
                          (const float*)d_in[36], (const float*)d_in[39]};
  const float* blst[4] = {(const float*)d_in[31], (const float*)d_in[34],
                          (const float*)d_in[37], (const float*)d_in[40]};

  float* ws = (float*)d_ws;
  const long XSZ = (long)kB * kS * kH;            // 1,572,864
  float* x    = ws;
  float* q    = ws + 1 * XSZ;
  float* kbuf = ws + 2 * XSZ;
  float* vbuf = ws + 3 * XSZ;
  float* ctx  = ws + 4 * XSZ;
  float* t1   = ws + 5 * XSZ;
  float* hff  = ws + 6 * XSZ;                     // B*S*FF = 4*XSZ
  // post-BERT reuse
  float* xt    = q;                               // [S,B,H]
  float* gxF   = vbuf;                            // [S,B,4LH] = 2,097,152
  float* gxB   = vbuf + (long)kS * kB * kG4;
  float* lstm1 = x;                               // [S,B,2LH] = 1,048,576
  float* lstm2 = gxB + (long)kS * kB * kG4;
  float* emit_sb = lstm2 + (long)kS * kB * kD2;   // [S,B,NT]
  int*   bp    = (int*)(emit_sb + (long)kS * kB * kNT);

  float* out_loss  = (float*)d_out;
  float* out_emit  = out_loss + 1;                       // [B,S,NT]
  float* out_preds = out_emit + (long)kB * kS * kNT;     // [B,S]

  // 1) embeddings + LN
  embed_ln_kernel<<<kRows, 256, 0, stream>>>(ids, word_emb, pos_emb, type_emb,
                                             emb_g, emb_b, x);

  // 2) BERT layers
  for (int l = 0; l < kNL; ++l) {
    const float* wql = wq + (long)l * kH * kH;  const float* bql = bq + (long)l * kH;
    const float* wkl = wk + (long)l * kH * kH;  const float* bkl = bk + (long)l * kH;
    const float* wvl = wv + (long)l * kH * kH;  const float* bvl = bv + (long)l * kH;
    const float* wol = wo + (long)l * kH * kH;  const float* bol = bo + (long)l * kH;
    const float* wil = wi + (long)l * kH * kFF; const float* bil = bi + (long)l * kFF;
    const float* wo2l = wo2 + (long)l * kFF * kH; const float* bo2l = bo2 + (long)l * kH;

    launch_gemm(x, wql, bql, q,    kRows, kH, kH, kH, kH, kH, 0, EPI_BIAS, stream);
    launch_gemm(x, wkl, bkl, kbuf, kRows, kH, kH, kH, kH, kH, 0, EPI_BIAS, stream);
    launch_gemm(x, wvl, bvl, vbuf, kRows, kH, kH, kH, kH, kH, 0, EPI_BIAS, stream);

    attention_kernel<<<(kB * kNH * kS) / 4, 128, 0, stream>>>(q, kbuf, vbuf, mask, ctx);

    launch_gemm(ctx, wol, bol, t1, kRows, kH, kH, kH, kH, kH, 0, EPI_BIAS, stream);
    ln_residual_kernel<<<kRows, 256, 0, stream>>>(x, t1,
        ln1g + (long)l * kH, ln1b + (long)l * kH);

    launch_gemm(x, wil, bil, hff, kRows, kFF, kH, kH, kFF, kFF, 0, EPI_BIAS_GELU, stream);
    launch_gemm(hff, wo2l, bo2l, t1, kRows, kH, kFF, kFF, kH, kH, 0, EPI_BIAS, stream);
    ln_residual_kernel<<<kRows, 256, 0, stream>>>(x, t1,
        ln2g + (long)l * kH, ln2b + (long)l * kH);
  }

  // 3) [B,S,H] -> [S,B,H]
  transpose01_kernel<<<1024, 256, 0, stream>>>(x, xt, kB, kS, kH);

  // 4) LSTM layer 0
  launch_gemm(xt, wih[0], blst[0], gxF, kRows, kG4, kH, kH, kH, kG4, 1, EPI_BIAS, stream);
  launch_gemm(xt, wih[1], blst[1], gxB, kRows, kG4, kH, kH, kH, kG4, 1, EPI_BIAS, stream);
  lstm_rec_kernel<<<2, 1024, 0, stream>>>(gxF, gxB, whh[0], whh[1], lstm1);

  // 5) LSTM layer 1
  launch_gemm(lstm1, wih[2], blst[2], gxF, kRows, kG4, kD2, kD2, kD2, kG4, 1, EPI_BIAS, stream);
  launch_gemm(lstm1, wih[3], blst[3], gxB, kRows, kG4, kD2, kD2, kD2, kG4, 1, EPI_BIAS, stream);
  lstm_rec_kernel<<<2, 1024, 0, stream>>>(gxF, gxB, whh[2], whh[3], lstm2);

  // 6) emission head: [S*B, 2LH] @ [2LH, NT]
  launch_gemm(lstm2, head_w, head_b, emit_sb, kRows, kNT, kD2, kD2, kNT, kNT,
              0, EPI_BIAS, stream);

  // 7) emit [S,B,NT] -> [B,S,NT] into output
  transpose01_kernel<<<64, 256, 0, stream>>>(emit_sb, out_emit, kS, kB, kNT);

  // 8) CRF loss
  crf_nll_kernel<<<1, 128, 0, stream>>>(emit_sb, labels, mask, trans, startv, endv,
                                        out_loss);

  // 9) Viterbi decode
  viterbi_kernel<<<1, 128, 0, stream>>>(emit_sb, trans, startv, endv, bp, out_preds);
}
